// GATModel_13804024889632
// MI455X (gfx1250) — compile-verified, matrix operand order checked by reading, not run on hardware
//
#include <hip/hip_runtime.h>
#include <math.h>

typedef __bf16 bf16;
typedef __attribute__((ext_vector_type(16))) __bf16 bf16x16;
typedef __attribute__((ext_vector_type(8)))  __bf16 bf16x8;
typedef __attribute__((ext_vector_type(8)))  float  f32x8;

#define N_NODES 50000
#define N_EDGES 800000
#define IN_FEATS 128
#define HID 64
#define HEADS 8
#define D1 (HEADS*HID)   /* 512 */
#define OUTF 64
#define NEG_SLOPE 0.2f

__device__ __forceinline__ void atomicMaxF(float* addr, float val) {
  // Sign-aware ordered-int trick: correct for mixed-sign floats.
  if (val >= 0.0f) atomicMax((int*)addr, __float_as_int(val));
  else             atomicMin((unsigned int*)addr, __float_as_uint(val));
}

// ---------------- conversion / transpose ----------------

__global__ void k_cvt_bf16(const float* __restrict__ s, bf16* __restrict__ d, long long n) {
  long long i = (long long)blockIdx.x * blockDim.x + threadIdx.x;
  if (i < n) d[i] = (bf16)s[i];
}

// W[K,N] fp32 -> Wt[N,K] bf16
__global__ void k_transpose_bf16(const float* __restrict__ W, bf16* __restrict__ Wt, int K, int N) {
  long long i = (long long)blockIdx.x * blockDim.x + threadIdx.x;
  if (i < (long long)K * N) {
    int k = (int)(i / N);
    int n = (int)(i % N);
    Wt[(long long)n * K + k] = (bf16)W[i];
  }
}

// ---------------- WMMA GEMM: C[M,Nout] = A[M,K] * B[K,Nout], B passed as Bt[Nout,K] ----------------
// block = 256 threads = 8 waves; blockIdx.y = M-tile; wave w owns N-tile blockIdx.x*8+w.
// K multiple of 32, K <= 512; M, Nout multiples of 16.
__global__ void k_gemm_wmma(const bf16* __restrict__ A, const bf16* __restrict__ Bt,
                            bf16* __restrict__ C, int M, int K, int Nout) {
  __shared__ bf16 As[16 * 512];   // 16KB of 320KB WGP LDS
  const int tileM = blockIdx.y;
  const int lane  = threadIdx.x & 31;
  const int wave  = threadIdx.x >> 5;
  const int r     = lane & 15;    // A row / B col / C col within tile
  const int h     = lane >> 4;    // lane half -> K-half select, C M-offset

  // Cooperative stage of the 16xK A tile into LDS (16B vector chunks).
  const int chunksPerRow = K >> 3;             // 8 bf16 = 16 bytes per chunk
  const int totalChunks  = 16 * chunksPerRow;
  for (int c = threadIdx.x; c < totalChunks; c += blockDim.x) {
    int row = c / chunksPerRow;
    int off = (c % chunksPerRow) << 3;
    *(bf16x8*)&As[row * K + off] =
        *(const bf16x8*)&A[((long long)tileM * 16 + row) * K + off];
  }
  __syncthreads();

  const int nt = blockIdx.x * 8 + wave;
  if (nt * 16 < Nout) {
    const bf16* bp = Bt + ((long long)(nt * 16 + r)) * K;
    f32x8 acc = {};
    for (int k0 = 0; k0 < K; k0 += 32) {
      // A frag: lane half h covers K = k0+8h..k0+8h+7 and k0+8h+16..k0+8h+23
      bf16x8 a0 = *(const bf16x8*)&As[r * K + k0 + h * 8];
      bf16x8 a1 = *(const bf16x8*)&As[r * K + k0 + h * 8 + 16];
      bf16x8 b0 = *(const bf16x8*)&bp[k0 + h * 8];
      bf16x8 b1 = *(const bf16x8*)&bp[k0 + h * 8 + 16];
      bf16x16 av, bv;
#pragma unroll
      for (int i = 0; i < 8; ++i) { av[i] = a0[i]; av[8 + i] = a1[i];
                                    bv[i] = b0[i]; bv[8 + i] = b1[i]; }
      acc = __builtin_amdgcn_wmma_f32_16x16x32_bf16(false, av, false, bv,
                                                    (short)0, acc, false, false);
    }
    // C layout: VGPR v -> M = v + 8*h ; N = lane%16
    const int ncol = nt * 16 + r;
#pragma unroll
    for (int v = 0; v < 8; ++v) {
      int m = tileM * 16 + v + 8 * h;
      C[(long long)m * Nout + ncol] = (bf16)acc[v];
    }
  }
}

// ---------------- layer-1 node prep: el/er dots + zero accumulators ----------------
__global__ void k_node_prep1(const bf16* __restrict__ h1, const float* __restrict__ al,
                             const float* __restrict__ ar, float* __restrict__ el,
                             float* __restrict__ er, float* __restrict__ emax,
                             float* __restrict__ esum, float* __restrict__ rst) {
  int i = blockIdx.x * blockDim.x + threadIdx.x;  // (node, head)
  if (i >= N_NODES * HEADS) return;
  int n = i >> 3, hd = i & 7;
  const bf16*  hp  = h1  + (long long)n * D1 + hd * HID;
  const float* alp = al  + hd * HID;
  const float* arp = ar  + hd * HID;
  float*       rp  = rst + (long long)n * D1 + hd * HID;
  float sl = 0.f, sr = 0.f;
  for (int d = 0; d < HID; ++d) {
    float v = (float)hp[d];
    sl += v * alp[d];
    sr += v * arp[d];
    rp[d] = 0.f;
  }
  el[i] = sl; er[i] = sr; emax[i] = -3.0e38f; esum[i] = 0.f;
}

// ---------------- layer-1 edge kernels ----------------
__global__ void k_edge_max1(const int* __restrict__ src, const int* __restrict__ dst,
                            const float* __restrict__ el, const float* __restrict__ er,
                            float* __restrict__ eatt, float* __restrict__ emax) {
  long long i = (long long)blockIdx.x * blockDim.x + threadIdx.x;
  if (i >= (long long)N_EDGES * HEADS) return;
  int e = (int)(i >> 3), hd = (int)(i & 7);
  int s = src[e], d = dst[e];
  float v = el[s * HEADS + hd] + er[d * HEADS + hd];
  v = v > 0.f ? v : NEG_SLOPE * v;
  eatt[i] = v;
  atomicMaxF(&emax[d * HEADS + hd], v);
}

__global__ void k_edge_exp1(const int* __restrict__ dst, float* __restrict__ eatt,
                            const float* __restrict__ emax, float* __restrict__ esum) {
  long long i = (long long)blockIdx.x * blockDim.x + threadIdx.x;
  if (i >= (long long)N_EDGES * HEADS) return;
  int e = (int)(i >> 3), hd = (int)(i & 7);
  int d = dst[e];
  float a = expf(eatt[i] - emax[d * HEADS + hd]);
  eatt[i] = a;
  atomicAdd(&esum[d * HEADS + hd], a);
}

__global__ void k_edge_agg1(const int* __restrict__ src, const int* __restrict__ dst,
                            const float* __restrict__ eatt, const float* __restrict__ esum,
                            const bf16* __restrict__ h1, float* __restrict__ rst) {
  long long i = (long long)blockIdx.x * blockDim.x + threadIdx.x;
  if (i >= (long long)N_EDGES * 64) return;
  int e  = (int)(i >> 6);
  int t  = (int)(i & 63);
  int hd = t >> 3;
  int c  = (t & 7) << 3;
  int s = src[e], d = dst[e];
  float coef = eatt[(long long)e * HEADS + hd] / esum[d * HEADS + hd];
  const bf16* hp = h1  + (long long)s * D1 + hd * HID + c;
  float*      rp = rst + (long long)d * D1 + hd * HID + c;
#pragma unroll
  for (int j = 0; j < 8; ++j) atomicAdd(&rp[j], coef * (float)hp[j]);
}

__global__ void k_finish1(const float* __restrict__ rst, const float* __restrict__ b1,
                          bf16* __restrict__ x1) {
  long long i = (long long)blockIdx.x * blockDim.x + threadIdx.x;
  if (i >= (long long)N_NODES * D1) return;
  float v = rst[i] + b1[i % D1];
  x1[i] = (bf16)fmaxf(v, 0.f);
}

// ---------------- layer-2 (1 head) ----------------
__global__ void k_node_prep2(const bf16* __restrict__ h2, const float* __restrict__ al2,
                             const float* __restrict__ ar2, float* __restrict__ el,
                             float* __restrict__ er, float* __restrict__ emax,
                             float* __restrict__ esum, float* __restrict__ rst) {
  int n = blockIdx.x * blockDim.x + threadIdx.x;
  if (n >= N_NODES) return;
  const bf16* hp = h2 + (long long)n * OUTF;
  float* rp = rst + (long long)n * OUTF;
  float sl = 0.f, sr = 0.f;
  for (int d = 0; d < OUTF; ++d) {
    float v = (float)hp[d];
    sl += v * al2[d];
    sr += v * ar2[d];
    rp[d] = 0.f;
  }
  el[n] = sl; er[n] = sr; emax[n] = -3.0e38f; esum[n] = 0.f;
}

__global__ void k_edge_max2(const int* __restrict__ src, const int* __restrict__ dst,
                            const float* __restrict__ el, const float* __restrict__ er,
                            float* __restrict__ eatt, float* __restrict__ emax) {
  int e = blockIdx.x * blockDim.x + threadIdx.x;
  if (e >= N_EDGES) return;
  float v = el[src[e]] + er[dst[e]];
  v = v > 0.f ? v : NEG_SLOPE * v;
  eatt[e] = v;
  atomicMaxF(&emax[dst[e]], v);
}

__global__ void k_edge_exp2(const int* __restrict__ dst, float* __restrict__ eatt,
                            const float* __restrict__ emax, float* __restrict__ esum) {
  int e = blockIdx.x * blockDim.x + threadIdx.x;
  if (e >= N_EDGES) return;
  int d = dst[e];
  float a = expf(eatt[e] - emax[d]);
  eatt[e] = a;
  atomicAdd(&esum[d], a);
}

__global__ void k_edge_agg2(const int* __restrict__ src, const int* __restrict__ dst,
                            const float* __restrict__ eatt, const float* __restrict__ esum,
                            const bf16* __restrict__ h2, float* __restrict__ rst) {
  long long i = (long long)blockIdx.x * blockDim.x + threadIdx.x;
  if (i >= (long long)N_EDGES * 8) return;
  int e = (int)(i >> 3);
  int c = (int)(i & 7) << 3;
  int s = src[e], d = dst[e];
  float coef = eatt[e] / esum[d];
  const bf16* hp = h2  + (long long)s * OUTF + c;
  float*      rp = rst + (long long)d * OUTF + c;
#pragma unroll
  for (int j = 0; j < 8; ++j) atomicAdd(&rp[j], coef * (float)hp[j]);
}

__global__ void k_finish2(const float* __restrict__ rst, const float* __restrict__ b2,
                          float* __restrict__ x2) {
  long long i = (long long)blockIdx.x * blockDim.x + threadIdx.x;
  if (i >= (long long)N_NODES * OUTF) return;
  x2[i] = fmaxf(rst[i] + b2[i % OUTF], 0.f);
}

// ---------------- edge prediction head ----------------
__global__ void k_edge_pred(const int* __restrict__ src, const int* __restrict__ dst,
                            const float* __restrict__ x2, const float* __restrict__ predW,
                            const float* __restrict__ predb, float* __restrict__ out) {
  int e = blockIdx.x * blockDim.x + threadIdx.x;
  if (e >= N_EDGES) return;
  const float* xs = x2 + (long long)src[e] * OUTF;
  const float* xd = x2 + (long long)dst[e] * OUTF;
  float z = predb[0];
  for (int j = 0; j < OUTF; ++j) z += xs[j] * predW[j];
  for (int j = 0; j < OUTF; ++j) z += xd[j] * predW[OUTF + j];
  out[e] = 1.f / (1.f + expf(-z));
}

// ---------------- host launch ----------------
static inline int nblk(long long n) { return (int)((n + 255) / 256); }

extern "C" void kernel_launch(void* const* d_in, const int* in_sizes, int n_in,
                              void* d_out, int out_size, void* d_ws, size_t ws_size,
                              hipStream_t stream) {
  (void)in_sizes; (void)n_in; (void)out_size; (void)ws_size;
  const int*   src   = (const int*)d_in[0];
  const int*   dst   = (const int*)d_in[1];
  /* d_in[2] = edge_type (unused, matches reference) */
  const float* feats = (const float*)d_in[3];
  const float* W1    = (const float*)d_in[4];
  const float* al1   = (const float*)d_in[5];
  const float* ar1   = (const float*)d_in[6];
  const float* b1    = (const float*)d_in[7];
  const float* W2    = (const float*)d_in[8];
  const float* al2   = (const float*)d_in[9];
  const float* ar2   = (const float*)d_in[10];
  const float* b2    = (const float*)d_in[11];
  const float* predW = (const float*)d_in[12];
  const float* predb = (const float*)d_in[13];
  float* out = (float*)d_out;

  char* ws = (char*)d_ws;
  size_t off = 0;
  auto alloc = [&](size_t bytes) -> void* {
    off = (off + 255) & ~(size_t)255;
    void* p = ws + off;
    off += bytes;
    return p;
  };

  bf16*  featA = (bf16*) alloc((size_t)N_NODES * IN_FEATS * 2);
  bf16*  w1t   = (bf16*) alloc((size_t)D1 * IN_FEATS * 2);
  bf16*  h1bf  = (bf16*) alloc((size_t)N_NODES * D1 * 2);
  float* el1   = (float*)alloc((size_t)N_NODES * HEADS * 4);
  float* er1   = (float*)alloc((size_t)N_NODES * HEADS * 4);
  float* emax1 = (float*)alloc((size_t)N_NODES * HEADS * 4);
  float* esum1 = (float*)alloc((size_t)N_NODES * HEADS * 4);
  float* eatt1 = (float*)alloc((size_t)N_EDGES * HEADS * 4);
  float* rst1  = (float*)alloc((size_t)N_NODES * D1 * 4);
  bf16*  x1bf  = (bf16*) alloc((size_t)N_NODES * D1 * 2);
  bf16*  w2t   = (bf16*) alloc((size_t)OUTF * D1 * 2);
  bf16*  h2bf  = (bf16*) alloc((size_t)N_NODES * OUTF * 2);
  float* el2   = (float*)alloc((size_t)N_NODES * 4);
  float* er2   = (float*)alloc((size_t)N_NODES * 4);
  float* emax2 = (float*)alloc((size_t)N_NODES * 4);
  float* esum2 = (float*)alloc((size_t)N_NODES * 4);
  float* eatt2 = (float*)alloc((size_t)N_EDGES * 4);
  float* rst2  = (float*)alloc((size_t)N_NODES * OUTF * 4);
  float* x2    = (float*)alloc((size_t)N_NODES * OUTF * 4);

  // ---- layer 1 ----
  k_cvt_bf16<<<nblk((long long)N_NODES * IN_FEATS), 256, 0, stream>>>(feats, featA, (long long)N_NODES * IN_FEATS);
  k_transpose_bf16<<<nblk((long long)IN_FEATS * D1), 256, 0, stream>>>(W1, w1t, IN_FEATS, D1);
  {
    dim3 g((D1 / 16 + 7) / 8, N_NODES / 16);
    k_gemm_wmma<<<g, 256, 0, stream>>>(featA, w1t, h1bf, N_NODES, IN_FEATS, D1);
  }
  k_node_prep1<<<nblk((long long)N_NODES * HEADS), 256, 0, stream>>>(h1bf, al1, ar1, el1, er1, emax1, esum1, rst1);
  k_edge_max1<<<nblk((long long)N_EDGES * HEADS), 256, 0, stream>>>(src, dst, el1, er1, eatt1, emax1);
  k_edge_exp1<<<nblk((long long)N_EDGES * HEADS), 256, 0, stream>>>(dst, eatt1, emax1, esum1);
  k_edge_agg1<<<nblk((long long)N_EDGES * 64), 256, 0, stream>>>(src, dst, eatt1, esum1, h1bf, rst1);
  k_finish1<<<nblk((long long)N_NODES * D1), 256, 0, stream>>>(rst1, b1, x1bf);

  // ---- layer 2 ----
  k_transpose_bf16<<<nblk((long long)D1 * OUTF), 256, 0, stream>>>(W2, w2t, D1, OUTF);
  {
    dim3 g((OUTF / 16 + 7) / 8, N_NODES / 16);
    k_gemm_wmma<<<g, 256, 0, stream>>>(x1bf, w2t, h2bf, N_NODES, D1, OUTF);
  }
  k_node_prep2<<<nblk(N_NODES), 256, 0, stream>>>(h2bf, al2, ar2, el2, er2, emax2, esum2, rst2);
  k_edge_max2<<<nblk(N_EDGES), 256, 0, stream>>>(src, dst, el2, er2, eatt2, emax2);
  k_edge_exp2<<<nblk(N_EDGES), 256, 0, stream>>>(dst, eatt2, emax2, esum2);
  k_edge_agg2<<<nblk((long long)N_EDGES * 8), 256, 0, stream>>>(src, dst, eatt2, esum2, h2bf, rst2);
  k_finish2<<<nblk((long long)N_NODES * OUTF), 256, 0, stream>>>(rst2, b2, x2);

  // ---- edge prediction ----
  k_edge_pred<<<nblk(N_EDGES), 256, 0, stream>>>(src, dst, x2, predW, predb, out);
}
